// GaddyMultiheadAttention_86990267613353
// MI455X (gfx1250) — matrix operand order repaired and checked by reading, not compiled, not run on hardware
//
#include <hip/hip_runtime.h>
#include <hip/hip_bf16.h>

typedef unsigned short u16;
typedef __attribute__((ext_vector_type(16))) __bf16 v16bf;
typedef __attribute__((ext_vector_type(8)))  float  v8f;

union BF16Frag {
    v16bf v;
    u16   u[16];
    uint4 q[2];        // two 16-byte halves: u[0..7], u[8..15]
};

__device__ __forceinline__ u16 f2bf(float f) {
    unsigned int x = __float_as_uint(f);
    x += 0x7fffu + ((x >> 16) & 1u);   // round-to-nearest-even
    return (u16)(x >> 16);
}

// ---------------------------------------------------------------------------
// Prep: convert weights to bf16, TRANSPOSED so B-fragments are contiguous:
//   wq_T/wk_T/wv_T : (n=h*d=768, k=e=768) row-major bf16
//   wo_T           : (n=e=768, k=h*d=768) row-major bf16
// ---------------------------------------------------------------------------
__global__ void prep_weights(const float* __restrict__ wq, const float* __restrict__ wk,
                             const float* __restrict__ wv, const float* __restrict__ wo,
                             u16* __restrict__ wq_T, u16* __restrict__ wk_T,
                             u16* __restrict__ wv_T, u16* __restrict__ wo_T) {
    int tid = blockIdx.x * blockDim.x + threadIdx.x;
    if (tid >= 768 * 768) return;
    int n  = tid / 768;                       // h*96 + d
    int e_ = tid % 768;
    int h  = n / 96;
    int d  = n % 96;
    int src = (h * 768 + e_) * 96 + d;        // w[h][e][d]
    wq_T[tid] = f2bf(wq[src]);
    wk_T[tid] = f2bf(wk[src]);
    wv_T[tid] = f2bf(wv[src]);
    wo_T[e_ * 768 + n] = f2bf(wo[tid]);       // wo flat is (h*d, e); store (e, h*d)
}

// emb_sum[h][i] = sum_d rel_emb[h][i][d][0]   -> (8, 199) fp32
__global__ void prep_embsum(const float* __restrict__ rel, float* __restrict__ embs) {
    int row = blockIdx.x * blockDim.x + threadIdx.x;
    if (row >= 8 * 199) return;
    const float* p = rel + (size_t)row * 96;
    float s = 0.f;
    for (int i = 0; i < 96; ++i) s += p[i];
    embs[row] = s;
}

// x (8192 x 768) fp32 -> bf16, vectorized
__global__ void x2bf(const float* __restrict__ x, u16* __restrict__ xb) {
    int t = blockIdx.x * blockDim.x + threadIdx.x;      // over 8192*768/4
    if (t >= 8192 * 768 / 4) return;
    float4 f = reinterpret_cast<const float4*>(x)[t];
    ushort4 v;
    v.x = f2bf(f.x); v.y = f2bf(f.y); v.z = f2bf(f.z); v.w = f2bf(f.w);
    reinterpret_cast<ushort4*>(xb)[t] = v;
}

// ---------------------------------------------------------------------------
// QKV projection GEMM, fully LDS-free: both fragments are direct b128 loads.
//   C(8192 x 768) = xb(8192 x 768, bf16) * W^T-fragments
// z selects Q/K/V. Q,K stored (b,h,s,d); V stored transposed (b,h,d,s).
// Block = 128 threads (4 waves). Wave tile: M=32 (2 frags) x N=64 (4 frags)
// -> 8 wmma per 12 b128 loads per K-step. Macro tile: M=128, N=64.
// ---------------------------------------------------------------------------
__global__ __launch_bounds__(128)
void qkv_gemm(const u16* __restrict__ xb,
              const u16* __restrict__ wT0, const u16* __restrict__ wT1, const u16* __restrict__ wT2,
              u16* __restrict__ o0, u16* __restrict__ o1, u16* __restrict__ o2) {
    const int z = blockIdx.z;
    const u16* wT  = (z == 0) ? wT0 : (z == 1) ? wT1 : wT2;
    u16*       out = (z == 0) ? o0  : (z == 1) ? o1  : o2;

    const int lane = threadIdx.x & 31;
    const int wave = threadIdx.x >> 5;
    const int half = lane >> 4;
    const int l16  = lane & 15;
    const int M0   = blockIdx.x * 128 + wave * 32;
    const int N0   = blockIdx.y * 64;

    v8f acc[2][4] = {};
    const u16* arow0 = xb + (size_t)(M0 + l16) * 768;
    const u16* arow1 = xb + (size_t)(M0 + 16 + l16) * 768;
    for (int k0 = 0; k0 < 768; k0 += 32) {
        BF16Frag a0, a1, b;
        a0.q[0] = *reinterpret_cast<const uint4*>(arow0 + k0 + half * 8);
        a0.q[1] = *reinterpret_cast<const uint4*>(arow0 + k0 + 16 + half * 8);
        a1.q[0] = *reinterpret_cast<const uint4*>(arow1 + k0 + half * 8);
        a1.q[1] = *reinterpret_cast<const uint4*>(arow1 + k0 + 16 + half * 8);
#pragma unroll
        for (int nt = 0; nt < 4; ++nt) {
            const u16* bp = wT + (size_t)(N0 + nt * 16 + l16) * 768 + k0 + half * 8;
            b.q[0] = *reinterpret_cast<const uint4*>(bp);
            b.q[1] = *reinterpret_cast<const uint4*>(bp + 16);
            acc[0][nt] = __builtin_amdgcn_wmma_f32_16x16x32_bf16(false, a0.v, false, b.v,
                                                                 (short)0, acc[0][nt], false, false);
            acc[1][nt] = __builtin_amdgcn_wmma_f32_16x16x32_bf16(false, a1.v, false, b.v,
                                                                 (short)0, acc[1][nt], false, false);
        }
    }

#pragma unroll
    for (int mt = 0; mt < 2; ++mt) {
#pragma unroll
        for (int nt = 0; nt < 4; ++nt) {
            const int n = N0 + nt * 16 + l16;
            const int h = n / 96, d = n % 96;
#pragma unroll
            for (int i = 0; i < 8; ++i) {
                int m  = i + 8 * half;
                int Mr = M0 + mt * 16 + m;
                int bb = Mr >> 11, srow = Mr & 2047;
                u16 val = f2bf(acc[mt][nt][i]);
                if (z == 2)   // V transposed: (b, h, d, s)
                    out[((size_t)((bb << 3) + h) * 96 + d) * 2048 + srow] = val;
                else          // Q, K: (b, h, s, d)
                    out[((size_t)((bb << 3) + h) * 2048 + srow) * 96 + d] = val;
            }
        }
    }
}

// ---------------------------------------------------------------------------
// Flash attention: Q,K bf16 (b,h,s,96), V bf16 transposed (b,h,96,s)
//   -> ctx bf16 (b, s, h*96)
// Block = 128 threads (4 waves); each wave owns 16 query rows; 32 keys/iter.
// All matrix fragments are direct b128 loads (global or wave-private LDS);
// the main loop has NO barriers.
// ---------------------------------------------------------------------------
__global__ __launch_bounds__(128)
void flash_attn(const u16* __restrict__ Q, const u16* __restrict__ K,
                const u16* __restrict__ VT, const float* __restrict__ embs,
                u16* __restrict__ ctx) {
    __shared__ __align__(16) u16 lds_p[4][16 * 40];    // per-wave P tile: [m][k], 80B rows
    __shared__ float lds_bias[200];

    const int lane = threadIdx.x & 31;
    const int wave = threadIdx.x >> 5;
    const int half = lane >> 4;
    const int l16  = lane & 15;
    const int hh   = blockIdx.y;
    const int bb   = blockIdx.z;
    const int q0   = blockIdx.x * 64 + wave * 16;

    const size_t headoff = (size_t)(bb * 8 + hh) * 2048 * 96;
    const u16* Qh  = Q + headoff;
    const u16* Kh  = K + headoff;
    const u16* VTh = VT + headoff;            // (96, 2048) within head

    for (int t = threadIdx.x; t < 199; t += 128) lds_bias[t] = embs[hh * 199 + t];

    // Persistent Q fragments: 3 chunks of K=32 over d=96, two b128 loads each
    BF16Frag qf[3];
    {
        const u16* qp = Qh + (size_t)(q0 + l16) * 96;
#pragma unroll
        for (int dc = 0; dc < 3; ++dc) {
            qf[dc].q[0] = *reinterpret_cast<const uint4*>(qp + dc * 32 + half * 8);
            qf[dc].q[1] = *reinterpret_cast<const uint4*>(qp + dc * 32 + 16 + half * 8);
        }
    }
    __syncthreads();                           // bias table visible

    float rowm[8], rowl[8];
#pragma unroll
    for (int i = 0; i < 8; ++i) { rowm[i] = -1e30f; rowl[i] = 0.f; }
    v8f acc[6] = {};
    const float scale = 0.10206207261596577f;  // 96^-0.5

    for (int kb = 0; kb < 2048; kb += 32) {
        if (kb + 32 < 2048) {                  // prefetch next K rows / V columns
            __builtin_prefetch(Kh + (size_t)(kb + 32 + lane) * 96, 0, 1);
            __builtin_prefetch(VTh + (size_t)(lane * 3) * 2048 + kb + 32, 0, 1);
        }

        // S = Q * K^T for two 16-key tiles (keys kb.. and kb+16..)
        v8f s0 = {}, s1 = {};
        {
            BF16Frag kf;
            const u16* kp0 = Kh + (size_t)(kb + l16) * 96;
#pragma unroll
            for (int dc = 0; dc < 3; ++dc) {
                kf.q[0] = *reinterpret_cast<const uint4*>(kp0 + dc * 32 + half * 8);
                kf.q[1] = *reinterpret_cast<const uint4*>(kp0 + dc * 32 + 16 + half * 8);
                s0 = __builtin_amdgcn_wmma_f32_16x16x32_bf16(false, qf[dc].v, false, kf.v,
                                                             (short)0, s0, false, false);
            }
            const u16* kp1 = Kh + (size_t)(kb + 16 + l16) * 96;
#pragma unroll
            for (int dc = 0; dc < 3; ++dc) {
                kf.q[0] = *reinterpret_cast<const uint4*>(kp1 + dc * 32 + half * 8);
                kf.q[1] = *reinterpret_cast<const uint4*>(kp1 + dc * 32 + 16 + half * 8);
                s1 = __builtin_amdgcn_wmma_f32_16x16x32_bf16(false, qf[dc].v, false, kf.v,
                                                             (short)0, s1, false, false);
            }
        }

        // scale + relative-position bias + online softmax
        float p0[8], p1[8];
#pragma unroll
        for (int i = 0; i < 8; ++i) {
            int m  = i + 8 * half;
            int q  = q0 + m;
            int r0 = kb + l16 - q;       r0 = r0 < -99 ? -99 : (r0 > 99 ? 99 : r0);
            int r1 = kb + 16 + l16 - q;  r1 = r1 < -99 ? -99 : (r1 > 99 ? 99 : r1);
            p0[i] = s0[i] * scale + lds_bias[r0 + 99];
            p1[i] = s1[i] * scale + lds_bias[r1 + 99];
        }
#pragma unroll
        for (int i = 0; i < 8; ++i) {
            float mx = fmaxf(p0[i], p1[i]);
#pragma unroll
            for (int off = 8; off >= 1; off >>= 1) mx = fmaxf(mx, __shfl_xor(mx, off, 16));
            float mnew = fmaxf(rowm[i], mx);
            float corr = __expf(rowm[i] - mnew);
            rowm[i] = mnew;
            p0[i] = __expf(p0[i] - mnew);
            p1[i] = __expf(p1[i] - mnew);
            float rs = p0[i] + p1[i];
#pragma unroll
            for (int off = 8; off >= 1; off >>= 1) rs += __shfl_xor(rs, off, 16);
            rowl[i] = rowl[i] * corr + rs;
#pragma unroll
            for (int nt = 0; nt < 6; ++nt) acc[nt][i] *= corr;
        }

        // C-frag (f32) -> A-frag (bf16) via wave-private LDS bounce (no barrier)
#pragma unroll
        for (int i = 0; i < 8; ++i) {
            int m = i + 8 * half;
            lds_p[wave][m * 40 + l16]      = f2bf(p0[i]);
            lds_p[wave][m * 40 + 16 + l16] = f2bf(p1[i]);
        }
        BF16Frag pf;
        {
            const u16* pp = &lds_p[wave][l16 * 40];
            pf.q[0] = *reinterpret_cast<const uint4*>(pp + half * 8);
            pf.q[1] = *reinterpret_cast<const uint4*>(pp + 16 + half * 8);
        }

        // ctx += P(16x32) * V(32x96): 6 N-tiles, B-frags direct from VT global
        BF16Frag vf;
#pragma unroll
        for (int nt = 0; nt < 6; ++nt) {
            const u16* vp = VTh + (size_t)(nt * 16 + l16) * 2048 + kb + half * 8;
            vf.q[0] = *reinterpret_cast<const uint4*>(vp);
            vf.q[1] = *reinterpret_cast<const uint4*>(vp + 16);
            acc[nt] = __builtin_amdgcn_wmma_f32_16x16x32_bf16(false, pf.v, false, vf.v,
                                                              (short)0, acc[nt], false, false);
        }
    }

    // normalize and write ctx (b, s, h*96) bf16
#pragma unroll
    for (int i = 0; i < 8; ++i) {
        int   m   = i + 8 * half;
        float inv = 1.f / rowl[i];
        size_t base = ((size_t)(bb * 2048 + q0 + m)) * 768 + hh * 96;
#pragma unroll
        for (int nt = 0; nt < 6; ++nt)
            ctx[base + nt * 16 + l16] = f2bf(acc[nt][i] * inv);
    }
}

// ---------------------------------------------------------------------------
// Output GEMM, LDS-free: out(8192 x 768, fp32) = ctx(8192 x 768, bf16) * w_o
// B-fragments from pre-transposed wo_T (e, h*d).
// Wave tile M=32 x N=64; macro tile M=128, N=64.
// ---------------------------------------------------------------------------
__global__ __launch_bounds__(128)
void out_gemm(const u16* __restrict__ A, const u16* __restrict__ BT,
              float* __restrict__ C) {
    const int lane = threadIdx.x & 31;
    const int wave = threadIdx.x >> 5;
    const int half = lane >> 4;
    const int l16  = lane & 15;
    const int M0   = blockIdx.x * 128 + wave * 32;
    const int N0   = blockIdx.y * 64;

    v8f acc[2][4] = {};
    const u16* arow0 = A + (size_t)(M0 + l16) * 768;
    const u16* arow1 = A + (size_t)(M0 + 16 + l16) * 768;
    for (int k0 = 0; k0 < 768; k0 += 32) {
        BF16Frag a0, a1, b;
        a0.q[0] = *reinterpret_cast<const uint4*>(arow0 + k0 + half * 8);
        a0.q[1] = *reinterpret_cast<const uint4*>(arow0 + k0 + 16 + half * 8);
        a1.q[0] = *reinterpret_cast<const uint4*>(arow1 + k0 + half * 8);
        a1.q[1] = *reinterpret_cast<const uint4*>(arow1 + k0 + 16 + half * 8);
#pragma unroll
        for (int nt = 0; nt < 4; ++nt) {
            const u16* bp = BT + (size_t)(N0 + nt * 16 + l16) * 768 + k0 + half * 8;
            b.q[0] = *reinterpret_cast<const uint4*>(bp);
            b.q[1] = *reinterpret_cast<const uint4*>(bp + 16);
            acc[0][nt] = __builtin_amdgcn_wmma_f32_16x16x32_bf16(false, a0.v, false, b.v,
                                                                 (short)0, acc[0][nt], false, false);
            acc[1][nt] = __builtin_amdgcn_wmma_f32_16x16x32_bf16(false, a1.v, false, b.v,
                                                                 (short)0, acc[1][nt], false, false);
        }
    }
#pragma unroll
    for (int mt = 0; mt < 2; ++mt) {
#pragma unroll
        for (int nt = 0; nt < 4; ++nt) {
#pragma unroll
            for (int i = 0; i < 8; ++i) {
                int m  = i + 8 * half;
                int Mr = M0 + mt * 16 + m;
                C[(size_t)Mr * 768 + N0 + nt * 16 + l16] = acc[mt][nt][i];
            }
        }
    }
}

// ---------------------------------------------------------------------------
extern "C" void kernel_launch(void* const* d_in, const int* in_sizes, int n_in,
                              void* d_out, int out_size, void* d_ws, size_t ws_size,
                              hipStream_t stream) {
    const float* x   = (const float*)d_in[0];   // (4, 2048, 768)
    const float* wq  = (const float*)d_in[1];   // (8, 768, 96)
    const float* wk  = (const float*)d_in[2];
    const float* wv  = (const float*)d_in[3];
    const float* wo  = (const float*)d_in[4];   // (8, 96, 768)
    const float* rel = (const float*)d_in[5];   // (8, 199, 96, 1)
    float* out = (float*)d_out;                 // (4, 2048, 768)

    char*  ws  = (char*)d_ws;
    size_t off = 0;
    auto carve = [&](size_t bytes) -> void* {
        off = (off + 255) & ~(size_t)255;
        void* p = ws + off;
        off += bytes;
        return p;
    };
    u16*   wq_T  = (u16*)carve((size_t)768 * 768 * 2);
    u16*   wk_T  = (u16*)carve((size_t)768 * 768 * 2);
    u16*   wv_T  = (u16*)carve((size_t)768 * 768 * 2);
    u16*   wo_T  = (u16*)carve((size_t)768 * 768 * 2);
    float* embs  = (float*)carve((size_t)8 * 199 * 4);
    u16*   xb    = (u16*)carve((size_t)8192 * 768 * 2);
    u16*   Qbf   = (u16*)carve((size_t)4 * 8 * 2048 * 96 * 2);
    u16*   Kbf   = (u16*)carve((size_t)4 * 8 * 2048 * 96 * 2);
    u16*   VTbf  = (u16*)carve((size_t)4 * 8 * 96 * 2048 * 2);
    u16*   ctxbf = (u16*)carve((size_t)4 * 2048 * 768 * 2);
    (void)ws_size; (void)in_sizes; (void)n_in; (void)out_size;

    hipLaunchKernelGGL(prep_weights, dim3((768 * 768 + 255) / 256), dim3(256), 0, stream,
                       wq, wk, wv, wo, wq_T, wk_T, wv_T, wo_T);
    hipLaunchKernelGGL(prep_embsum, dim3((8 * 199 + 255) / 256), dim3(256), 0, stream,
                       rel, embs);
    hipLaunchKernelGGL(x2bf, dim3((8192 * 768 / 4 + 255) / 256), dim3(256), 0, stream,
                       x, xb);
    hipLaunchKernelGGL(qkv_gemm, dim3(64, 12, 3), dim3(128), 0, stream,
                       xb, wq_T, wk_T, wv_T, Qbf, Kbf, VTbf);
    hipLaunchKernelGGL(flash_attn, dim3(32, 8, 4), dim3(128), 0, stream,
                       Qbf, Kbf, VTbf, embs, ctxbf);
    hipLaunchKernelGGL(out_gemm, dim3(64, 12), dim3(128), 0, stream,
                       ctxbf, wo_T, out);
}
